// MultiHeadAttention_70368744177978
// MI455X (gfx1250) — matrix operand order repaired and checked by reading, not compiled, not run on hardware
//
#include <hip/hip_runtime.h>

// ---------------- types ----------------
typedef __attribute__((ext_vector_type(16))) __bf16 v16bf;
typedef __attribute__((ext_vector_type(8)))  float  v8f;
typedef __attribute__((ext_vector_type(4)))  int    i4;

typedef __attribute__((address_space(1))) i4 GI4;   // global int4
typedef __attribute__((address_space(3))) i4 LI4;   // LDS int4

union Frag {
    v16bf v;
    uint4 q[2];
};

__device__ __forceinline__ v8f zero8() {
    v8f z = {0.f, 0.f, 0.f, 0.f, 0.f, 0.f, 0.f, 0.f};
    return z;
}

__device__ __forceinline__ unsigned short f2bf(float f) {
    unsigned int u = __float_as_uint(f);
    unsigned int r = u + 0x7FFFu + ((u >> 16) & 1u);   // round-to-nearest-even
    return (unsigned short)(r >> 16);
}

// ---------------- CDNA5 async global->LDS copy ----------------
#if defined(__has_builtin)
#  if __has_builtin(__builtin_amdgcn_global_load_async_to_lds_b128)
#    define HAVE_ASYNC_LDS 1
#  endif
#endif

#ifdef HAVE_ASYNC_LDS
#  if __has_builtin(__builtin_amdgcn_s_wait_asynccnt)
#    define WAIT_ASYNC(n) __builtin_amdgcn_s_wait_asynccnt(n)
#  else
#    define WAIT_ASYNC(n) asm volatile("s_wait_asynccnt %0" :: "i"(n) : "memory")
#  endif
#else
#  define WAIT_ASYNC(n) do {} while (0)
#endif

__device__ __forceinline__ void cp16_async(unsigned short* l, const unsigned short* g) {
#ifdef HAVE_ASYNC_LDS
    __builtin_amdgcn_global_load_async_to_lds_b128((GI4*)(size_t)g, (LI4*)l, 0, 0);
#else
    *(uint4*)l = *(const uint4*)g;
#endif
}

// ---------------- fp32 -> bf16 conversion ----------------
__global__ __launch_bounds__(256)
void cvt_f32_bf16x4(const float4* __restrict__ src, ushort4* __restrict__ dst, size_t n4) {
    size_t i = (size_t)blockIdx.x * blockDim.x + threadIdx.x;
    size_t stride = (size_t)gridDim.x * blockDim.x;
    for (; i < n4; i += stride) {
        float4 v = src[i];
        ushort4 r;
        r.x = f2bf(v.x); r.y = f2bf(v.y); r.z = f2bf(v.z); r.w = f2bf(v.w);
        dst[i] = r;
    }
}

// ---------------- NT GEMM: Y[M,N] = scale * (X[M,K] @ W[N,K]^T) (+bias) ----------------
// block = 256 threads (8 waves), 128x128 output tile, double-buffered 64-wide K slabs.
#define GTK  64
#define GLDA 72   // 64 + 8 pad (ushorts): 16B-aligned rows, bank-spread

// stage one 128x64 slab of X and W into LDS (async when available)
__device__ __forceinline__ void gemm_stage(const unsigned short* __restrict__ X,
                                           const unsigned short* __restrict__ W,
                                           unsigned short* dA, unsigned short* dB,
                                           int m0, int n0, int K, int k0, int tid)
{
#ifdef HAVE_ASYNC_LDS
    #pragma unroll
    for (int i = 0; i < 4; ++i) {
        int id = tid + i * 256;         // 0..1023
        int r  = id >> 3;               // 128 rows
        int c  = (id & 7) * 8;          // 8 chunks of 8 bf16
        cp16_async(&dA[r * GLDA + c], X + (size_t)(m0 + r) * K + k0 + c);
        cp16_async(&dB[r * GLDA + c], W + (size_t)(n0 + r) * K + k0 + c);
    }
#else
    uint4 tA[4], tB[4];
    #pragma unroll
    for (int i = 0; i < 4; ++i) {       // batch loads first (avoid per-chunk stalls)
        int id = tid + i * 256;
        int r  = id >> 3;
        int c  = (id & 7) * 8;
        tA[i] = *(const uint4*)(X + (size_t)(m0 + r) * K + k0 + c);
        tB[i] = *(const uint4*)(W + (size_t)(n0 + r) * K + k0 + c);
    }
    #pragma unroll
    for (int i = 0; i < 4; ++i) {
        int id = tid + i * 256;
        int r  = id >> 3;
        int c  = (id & 7) * 8;
        *(uint4*)&dA[r * GLDA + c] = tA[i];
        *(uint4*)&dB[r * GLDA + c] = tB[i];
    }
#endif
}

__global__ __launch_bounds__(256)
void gemm_nt_bf16(const unsigned short* __restrict__ X,
                  const unsigned short* __restrict__ W,
                  unsigned short* __restrict__ Yb,     // bf16 out (if !f32out)
                  float* __restrict__ Yf,              // f32 out (if f32out)
                  const float* __restrict__ bias,      // nullable
                  int M, int N, int K, float scale, int f32out)
{
    __shared__ unsigned short sA[2][128 * GLDA];
    __shared__ unsigned short sB[2][128 * GLDA];

    const int tid  = threadIdx.x;
    const int w    = tid >> 5;
    const int lane = tid & 31;
    const int half = lane >> 4;
    const int l16  = lane & 15;

    const int m0 = blockIdx.y * 128;
    const int n0 = blockIdx.x * 128;
    const int rw = (w & 3) * 32;   // wave row offset (2 m-tiles)
    const int cw = (w >> 2) * 64;  // wave col offset (4 n-tiles)

    v8f acc[2][4];
    for (int mt = 0; mt < 2; ++mt)
        for (int nt = 0; nt < 4; ++nt)
            acc[mt][nt] = zero8();

    const int nslab = K / GTK;
    gemm_stage(X, W, sA[0], sB[0], m0, n0, K, 0, tid);

    for (int s = 0; s < nslab; ++s) {
        const int cur = s & 1;
        if (s + 1 < nslab) {
            gemm_stage(X, W, sA[cur ^ 1], sB[cur ^ 1], m0, n0, K, (s + 1) * GTK, tid);
            WAIT_ASYNC(8);   // 8 async ops in flight for next slab; current slab done
        } else {
            WAIT_ASYNC(0);
        }
        __syncthreads();

        const unsigned short* a = sA[cur];
        const unsigned short* bS = sB[cur];
        #pragma unroll
        for (int kk = 0; kk < 2; ++kk) {
            Frag aF[2], bF[4];
            #pragma unroll
            for (int mt = 0; mt < 2; ++mt) {
                int r  = rw + mt * 16 + l16;
                int kA = kk * 32 + half * 8;
                aF[mt].q[0] = *(const uint4*)&a[r * GLDA + kA];
                aF[mt].q[1] = *(const uint4*)&a[r * GLDA + kA + 16];
            }
            #pragma unroll
            for (int nt = 0; nt < 4; ++nt) {
                int r  = cw + nt * 16 + l16;
                int kB = kk * 32 + half * 8;
                bF[nt].q[0] = *(const uint4*)&bS[r * GLDA + kB];
                bF[nt].q[1] = *(const uint4*)&bS[r * GLDA + kB + 16];
            }
            #pragma unroll
            for (int mt = 0; mt < 2; ++mt)
                #pragma unroll
                for (int nt = 0; nt < 4; ++nt)
                    acc[mt][nt] = __builtin_amdgcn_wmma_f32_16x16x32_bf16(
                        false, aF[mt].v, false, bF[nt].v, (short)0, acc[mt][nt], false, false);
        }
        __syncthreads();   // all waves done with 'cur' before it is restaged
    }

    // epilogue: C/D layout — VGPR j holds row (j + 8*half), lanes 0..15 hold cols
    for (int mt = 0; mt < 2; ++mt)
        for (int nt = 0; nt < 4; ++nt)
            #pragma unroll
            for (int j = 0; j < 8; ++j) {
                int row = m0 + rw + mt * 16 + j + half * 8;
                int col = n0 + cw + nt * 16 + l16;
                float v = acc[mt][nt][j] * scale;
                if (f32out)
                    Yf[(size_t)row * N + col] = v + (bias ? bias[col] : 0.f);
                else
                    Yb[(size_t)row * N + col] = f2bf(v);
            }
}

// ---------------- flash attention ----------------
// grid = (T/128, H, B), block = 256 (8 waves; each wave owns 16 query rows).
// Q/K/V/Ctx layout: [B, T, H*D] bf16 (head-major inner), D = 128.
// Double-buffered K/V tiles; K tile via async LDS copy; V transposed manually.
#define FLDK 136  // 128 + 8 pad ushorts

__global__ __launch_bounds__(256)
void flash_attn(const unsigned short* __restrict__ Q,
                const unsigned short* __restrict__ Kb,
                const unsigned short* __restrict__ Vb,
                const int* __restrict__ pm,
                unsigned short* __restrict__ Ctx,
                int B, int T, int H, int Dh)
{
    __shared__ unsigned short sK[2][128 * FLDK];  // K tile; reused for P after S computed
    __shared__ unsigned short sV[2][128 * FLDK];  // transposed V tile: sV[buf][d * FLDK + s]

    const int tid  = threadIdx.x;
    const int w    = tid >> 5;
    const int lane = tid & 31;
    const int half = lane >> 4;
    const int l16  = lane & 15;

    const int qb0 = blockIdx.x * 128;
    const int h   = blockIdx.y;
    const int b   = blockIdx.z;
    const int C   = H * Dh;

    const unsigned short* qbase = Q  + ((size_t)b * T) * C + (size_t)h * Dh;
    const unsigned short* kbase = Kb + ((size_t)b * T) * C + (size_t)h * Dh;
    const unsigned short* vbase = Vb + ((size_t)b * T) * C + (size_t)h * Dh;

    const int qr = qb0 + w * 16;

    // preload this wave's Q fragments: 16 rows x 128 (K-dim) -> 4 A-fragments
    Frag qF[4];
    {
        const unsigned short* qrow = qbase + (size_t)(qr + l16) * C;
        #pragma unroll
        for (int kk = 0; kk < 4; ++kk) {
            int off = kk * 32 + half * 8;
            qF[kk].q[0] = *(const uint4*)(qrow + off);
            qF[kk].q[1] = *(const uint4*)(qrow + off + 16);
        }
    }

    // per-row running softmax state (rows j + 8*half, matching C-layout)
    float mrow[8], lrow[8], rowKeep[8];
    int rowIdx[8];
    #pragma unroll
    for (int j = 0; j < 8; ++j) {
        rowIdx[j]  = qr + j + half * 8;
        rowKeep[j] = pm[b * T + rowIdx[j]] ? 1.f : 0.f;
        mrow[j] = -3.0e38f;
        lrow[j] = 0.f;
    }
    v8f cacc[8];
    #pragma unroll
    for (int dt = 0; dt < 8; ++dt) cacc[dt] = zero8();

    const int nkb = (int)blockIdx.x + 1;

    // stage helper: K async, V transposed via regs
    auto stage_kv = [&](int buf, int s0) {
        unsigned short* dK = sK[buf];
        unsigned short* dV = sV[buf];
        #pragma unroll
        for (int i = 0; i < 8; ++i) {
            int id = tid + i * 256;
            int r  = id >> 4;
            int c8 = (id & 15) * 8;
            cp16_async(&dK[r * FLDK + c8], kbase + (size_t)(s0 + r) * C + c8);
        }
        uint4 tV[8];
        #pragma unroll
        for (int i = 0; i < 8; ++i) {
            int id = tid + i * 256;
            int r  = id >> 4;
            int c8 = (id & 15) * 8;
            tV[i] = *(const uint4*)(vbase + (size_t)(s0 + r) * C + c8);
        }
        #pragma unroll
        for (int i = 0; i < 8; ++i) {
            int id = tid + i * 256;
            int r  = id >> 4;
            int c8 = (id & 15) * 8;
            const unsigned short* ve = (const unsigned short*)&tV[i];
            #pragma unroll
            for (int e = 0; e < 8; ++e) dV[(c8 + e) * FLDK + r] = ve[e];
        }
    };

    stage_kv(0, 0);

    for (int kb = 0; kb < nkb; ++kb) {
        const int cur = kb & 1;
        const int s0  = kb * 128;
        if (kb + 1 < nkb) {
            stage_kv(cur ^ 1, (kb + 1) * 128);
            WAIT_ASYNC(8);
        } else {
            WAIT_ASYNC(0);
        }
        __syncthreads();

        const unsigned short* kT = sK[cur];
        const unsigned short* vT = sV[cur];
        unsigned short* pT = sK[cur];

        // S = Q K^T : 8 key tiles of 16, contraction over D in 4 steps of 32
        v8f st[8];
        #pragma unroll
        for (int nt = 0; nt < 8; ++nt) {
            v8f c = zero8();
            #pragma unroll
            for (int kk = 0; kk < 4; ++kk) {
                Frag bK;
                int srow = nt * 16 + l16;
                int off  = kk * 32 + half * 8;
                bK.q[0] = *(const uint4*)&kT[srow * FLDK + off];
                bK.q[1] = *(const uint4*)&kT[srow * FLDK + off + 16];
                c = __builtin_amdgcn_wmma_f32_16x16x32_bf16(
                        false, qF[kk].v, false, bK.v, (short)0, c, false, false);
            }
            st[nt] = c;
        }
        __syncthreads();  // all waves done reading K region before it is reused for P

        // masks (padding masks query rows; causal by index)
        #pragma unroll
        for (int nt = 0; nt < 8; ++nt) {
            int col = s0 + nt * 16 + l16;
            #pragma unroll
            for (int j = 0; j < 8; ++j) {
                float s = st[nt][j];
                s = rowKeep[j] > 0.f ? s : -1.0e9f;
                s = (col <= rowIdx[j]) ? s : -3.0e38f;
                st[nt][j] = s;
            }
        }
        // row max across 8 tiles, then across the 16 lanes holding the row
        float lm[8];
        #pragma unroll
        for (int j = 0; j < 8; ++j) {
            float v = st[0][j];
            #pragma unroll
            for (int nt = 1; nt < 8; ++nt) v = fmaxf(v, st[nt][j]);
            lm[j] = v;
        }
        for (int off = 1; off < 16; off <<= 1)
            #pragma unroll
            for (int j = 0; j < 8; ++j)
                lm[j] = fmaxf(lm[j], __shfl_xor(lm[j], off, 32));

        float alpha[8];
        #pragma unroll
        for (int j = 0; j < 8; ++j) {
            float mnew = fmaxf(mrow[j], lm[j]);
            alpha[j] = __expf(mrow[j] - mnew);
            mrow[j]  = mnew;
            lrow[j] *= alpha[j];
        }
        #pragma unroll
        for (int dt = 0; dt < 8; ++dt)
            #pragma unroll
            for (int j = 0; j < 8; ++j)
                cacc[dt][j] *= alpha[j];

        // P = exp(S - m), accumulate row sums, spill P (bf16) into this wave's LDS rows
        float ls[8];
        #pragma unroll
        for (int j = 0; j < 8; ++j) ls[j] = 0.f;
        #pragma unroll
        for (int nt = 0; nt < 8; ++nt) {
            #pragma unroll
            for (int j = 0; j < 8; ++j) {
                float p = __expf(st[nt][j] - mrow[j]);
                ls[j] += p;
                pT[(w * 16 + j + half * 8) * FLDK + nt * 16 + l16] = f2bf(p);
            }
        }
        for (int off = 1; off < 16; off <<= 1)
            #pragma unroll
            for (int j = 0; j < 8; ++j)
                ls[j] += __shfl_xor(ls[j], off, 32);
        #pragma unroll
        for (int j = 0; j < 8; ++j) lrow[j] += ls[j];
        __syncthreads();

        // ctx += P @ V  (contraction over s in 4 steps of 32; V is transposed in LDS)
        #pragma unroll
        for (int kk = 0; kk < 4; ++kk) {
            Frag aP;
            int prow = w * 16 + l16;
            int aoff = kk * 32 + half * 8;
            aP.q[0] = *(const uint4*)&pT[prow * FLDK + aoff];
            aP.q[1] = *(const uint4*)&pT[prow * FLDK + aoff + 16];
            #pragma unroll
            for (int dt = 0; dt < 8; ++dt) {
                Frag bV;
                int drow = dt * 16 + l16;
                int boff = kk * 32 + half * 8;
                bV.q[0] = *(const uint4*)&vT[drow * FLDK + boff];
                bV.q[1] = *(const uint4*)&vT[drow * FLDK + boff + 16];
                cacc[dt] = __builtin_amdgcn_wmma_f32_16x16x32_bf16(
                               false, aP.v, false, bV.v, (short)0, cacc[dt], false, false);
            }
        }
        __syncthreads();   // before next iteration restages this buffer pair
    }

    // normalize and write ctx (bf16)
    #pragma unroll
    for (int dt = 0; dt < 8; ++dt)
        #pragma unroll
        for (int j = 0; j < 8; ++j) {
            float v = cacc[dt][j] / lrow[j];
            Ctx[((size_t)b * T + rowIdx[j]) * C + (size_t)h * Dh + dt * 16 + l16] = f2bf(v);
        }
}

// ---------------- host launcher ----------------
extern "C" void kernel_launch(void* const* d_in, const int* in_sizes, int n_in,
                              void* d_out, int out_size, void* d_ws, size_t ws_size,
                              hipStream_t stream) {
    (void)in_sizes; (void)n_in; (void)out_size; (void)ws_size;

    constexpr int Bc = 2, Tc = 2048, Cc = 2048, Hc = 16, Dc = 128;
    constexpr int M = Bc * Tc;                 // 4096
    const float qscale = 0.08838834764831845f; // D^-0.5

    const float* x  = (const float*)d_in[0];
    const int*   pm = (const int*)d_in[1];
    const float* Wq = (const float*)d_in[2];
    const float* Wk = (const float*)d_in[3];
    const float* Wv = (const float*)d_in[4];
    const float* Wp = (const float*)d_in[5];
    const float* bp = (const float*)d_in[6];
    float* out = (float*)d_out;

    // workspace carve-up (bf16 elements)
    unsigned short* xb  = (unsigned short*)d_ws;
    unsigned short* wqb = xb  + (size_t)M * Cc;
    unsigned short* wkb = wqb + (size_t)Cc * Cc;
    unsigned short* wvb = wkb + (size_t)Cc * Cc;
    unsigned short* wpb = wvb + (size_t)Cc * Cc;
    unsigned short* qb  = wpb + (size_t)Cc * Cc;
    unsigned short* kb  = qb  + (size_t)M * Cc;
    unsigned short* vb  = kb  + (size_t)M * Cc;
    unsigned short* cb  = vb  + (size_t)M * Cc;

    // 1) fp32 -> bf16 conversions
    cvt_f32_bf16x4<<<1024, 256, 0, stream>>>((const float4*)x,  (ushort4*)xb,  (size_t)M * Cc / 4);
    cvt_f32_bf16x4<<<512,  256, 0, stream>>>((const float4*)Wq, (ushort4*)wqb, (size_t)Cc * Cc / 4);
    cvt_f32_bf16x4<<<512,  256, 0, stream>>>((const float4*)Wk, (ushort4*)wkb, (size_t)Cc * Cc / 4);
    cvt_f32_bf16x4<<<512,  256, 0, stream>>>((const float4*)Wv, (ushort4*)wvb, (size_t)Cc * Cc / 4);
    cvt_f32_bf16x4<<<512,  256, 0, stream>>>((const float4*)Wp, (ushort4*)wpb, (size_t)Cc * Cc / 4);

    // 2) projections: Q (scaled by D^-0.5), K, V
    dim3 ggrid(Cc / 128, M / 128);
    gemm_nt_bf16<<<ggrid, 256, 0, stream>>>(xb, wqb, qb, nullptr, nullptr, M, Cc, Cc, qscale, 0);
    gemm_nt_bf16<<<ggrid, 256, 0, stream>>>(xb, wkb, kb, nullptr, nullptr, M, Cc, Cc, 1.0f, 0);
    gemm_nt_bf16<<<ggrid, 256, 0, stream>>>(xb, wvb, vb, nullptr, nullptr, M, Cc, Cc, 1.0f, 0);

    // 3) flash attention -> ctx (bf16)
    flash_attn<<<dim3(Tc / 128, Hc, Bc), 256, 0, stream>>>(qb, kb, vb, pm, cb, Bc, Tc, Hc, Dc);

    // 4) output projection with bias -> fp32 d_out
    gemm_nt_bf16<<<ggrid, 256, 0, stream>>>(cb, wpb, nullptr, out, bp, M, Cc, Cc, 1.0f, 1);
}
